// DistanceSVM_31550829756879
// MI455X (gfx1250) — matrix operand order repaired and checked by reading, not compiled, not run on hardware
//
#include <hip/hip_runtime.h>

// DistanceSVM for MI455X (gfx1250, wave32):
//   out[n] = max_avg_distance - sum_c w[c] * sqrt(max(||x_n||^2 + ||c||^2 - 2 x_n.c, 0))
// GEMM done with v_wmma_f32_16x16x32_bf16 using bf16 hi/lo split (3-term) for ~fp32 accuracy.
// sqrt via raw v_sqrt_f32 (no denormal fixup sequence) to keep inner-loop VALU lean.

typedef __attribute__((ext_vector_type(16))) __bf16 v16bf;
typedef __attribute__((ext_vector_type(8)))  float  v8f;

#define N_TOTAL 131072
#define C_TOTAL 1024
#define D_DIM   64

// ---------------- Prep kernel: one block. ----------------
// Converts centers (C x 64 fp32) -> bf16 hi/lo planes, computes ||c||^2 and
// normalized weights w[c] = |coef|/sum|coef| (or |coef| if sum==0).
__global__ __launch_bounds__(256) void svm_prep_kernel(
    const float* __restrict__ centers, const float* __restrict__ coefs,
    float2* __restrict__ cw, __bf16* __restrict__ chi, __bf16* __restrict__ clo) {
  __shared__ float red[256];
  const int t = threadIdx.x;
  float asum = 0.0f;
  float c2v[4];
#pragma unroll
  for (int i = 0; i < 4; ++i) {
    const int c = t + i * 256;
    const float* cr = centers + (size_t)c * D_DIM;
    float c2 = 0.0f;
#pragma unroll
    for (int d = 0; d < D_DIM; ++d) {
      float v = cr[d];
      c2 += v * v;
      __bf16 h = (__bf16)v;
      chi[(size_t)c * D_DIM + d] = h;
      clo[(size_t)c * D_DIM + d] = (__bf16)(v - (float)h);
    }
    c2v[i] = c2;
    asum += fabsf(coefs[c]);
  }
  red[t] = asum;
  __syncthreads();
  for (int s = 128; s > 0; s >>= 1) {
    if (t < s) red[t] += red[t + s];
    __syncthreads();
  }
  const float s = red[0];
  const float inv = (s == 0.0f) ? 1.0f : (1.0f / s);
#pragma unroll
  for (int i = 0; i < 4; ++i) {
    const int c = t + i * 256;
    cw[c] = make_float2(c2v[i], fabsf(coefs[c]) * inv);
  }
}

// ---------------- Main kernel ----------------
// 256 threads = 8 waves; each wave owns a 16-row tile of x (held as bf16 hi/lo
// A-matrix registers) and sweeps all 1024 centers in chunks of 16 columns.
__global__ __launch_bounds__(256) void svm_main_kernel(
    const float* __restrict__ x, const float2* __restrict__ cw,
    const __bf16* __restrict__ chi, const __bf16* __restrict__ clo,
    const float* __restrict__ mad, float* __restrict__ out) {
  const int lane    = threadIdx.x & 31;
  const int waveid  = threadIdx.x >> 5;
  const int rowbase = blockIdx.x * 128 + waveid * 16;
  const int m       = lane & 15;            // row within tile (A) / column (B)
  const bool hihalf = (lane & 16) != 0;
  const int row     = rowbase + m;

  // ---- Load A tile (x rows) as bf16 hi/lo; accumulate ||x||^2 partials ----
  // A k order used here (16x32): lanes 0-15 row m: elems 0..15 -> K 0..15 (contiguous)
  //                           lanes 16-31 row m: K offset +16.
  v16bf a_hi[2], a_lo[2];
  float x2p = 0.0f;
  const float* xr = x + (size_t)row * D_DIM;
#pragma unroll
  for (int ks = 0; ks < 2; ++ks) {
    const int f0 = ks * 32 + (hihalf ? 16 : 0);
#pragma unroll
    for (int i = 0; i < 8; ++i) {
      float v0 = xr[f0 + i];
      float v1 = xr[f0 + 8 + i];
      x2p += v0 * v0 + v1 * v1;
      __bf16 h0 = (__bf16)v0;
      __bf16 h1 = (__bf16)v1;
      a_hi[ks][i]     = h0;
      a_lo[ks][i]     = (__bf16)(v0 - (float)h0);
      a_hi[ks][8 + i] = h1;
      a_lo[ks][8 + i] = (__bf16)(v1 - (float)h1);
    }
  }
  // Lanes m and m+16 cover complementary K halves of row m: combine.
  x2p += __shfl_xor(x2p, 16);
  // Broadcast per-row ||x||^2 into the D-matrix layout: VGPR v, this lane -> row v (+8 hi half).
  float x2r[8];
#pragma unroll
  for (int v = 0; v < 8; ++v)
    x2r[v] = __shfl(x2p, v + (hihalf ? 8 : 0));

  float wsum[8] = {0, 0, 0, 0, 0, 0, 0, 0};

  // ---- Sweep centers: 64 chunks of 16 columns ----
  for (int cb = 0; cb < C_TOTAL; cb += 16) {
    const int c = cb + m;  // this lane's column (center index)
    // B k order used here: lane c holds column c, K = koff + (lane<16 ? 0 : 16) + 0..15, contiguous.
    const __bf16* bh = chi + (size_t)c * D_DIM;
    const __bf16* bl = clo + (size_t)c * D_DIM;

    v8f acc = {0, 0, 0, 0, 0, 0, 0, 0};
#pragma unroll
    for (int ks = 0; ks < 2; ++ks) {
      const int g0 = ks * 32 + (hihalf ? 16 : 0);
      v16bf b_hi = *(const v16bf*)(bh + g0);
      v16bf b_lo = *(const v16bf*)(bl + g0);
      // dot ~= hi*hi + hi*lo + lo*hi  (fp32-accurate split product)
      acc = __builtin_amdgcn_wmma_f32_16x16x32_bf16(false, a_hi[ks], false, b_hi,
                                                    (short)0, acc, false, false);
      acc = __builtin_amdgcn_wmma_f32_16x16x32_bf16(false, a_hi[ks], false, b_lo,
                                                    (short)0, acc, false, false);
      acc = __builtin_amdgcn_wmma_f32_16x16x32_bf16(false, a_lo[ks], false, b_hi,
                                                    (short)0, acc, false, false);
    }

    // Epilogue for this 16x16 tile: element (v, lane) = (row x2r[v], col c).
    const float2 cwv = cw[c];  // {||center||^2, weight}
#pragma unroll
    for (int v = 0; v < 8; ++v) {
      float d2 = x2r[v] + cwv.x - 2.0f * acc[v];
      d2 = fmaxf(d2, 0.0f);
      // Raw v_sqrt_f32: inputs are O(1), never denormal -> skip libm fixup sequence.
      wsum[v] += cwv.y * __builtin_amdgcn_sqrtf(d2);
    }
  }

  // ---- Per-row reduction across the 16 column-lanes of each half ----
  const float madv = mad[0];
#pragma unroll
  for (int v = 0; v < 8; ++v) {
    float s = wsum[v];
    s += __shfl_xor(s, 1);
    s += __shfl_xor(s, 2);
    s += __shfl_xor(s, 4);
    s += __shfl_xor(s, 8);
    if (m == 0) {  // lanes 0 and 16 hold row v and row v+8 totals
      out[rowbase + v + (hihalf ? 8 : 0)] = madv - s;
    }
  }
}

extern "C" void kernel_launch(void* const* d_in, const int* in_sizes, int n_in,
                              void* d_out, int out_size, void* d_ws, size_t ws_size,
                              hipStream_t stream) {
  const float* x       = (const float*)d_in[0];  // (N, 64) fp32
  const float* centers = (const float*)d_in[1];  // (1024, 64) fp32
  const float* coefs   = (const float*)d_in[2];  // (1024,) fp32
  const float* mad     = (const float*)d_in[3];  // (1,) fp32
  float* out           = (float*)d_out;          // (N,) fp32

  // Workspace layout (all 32B-aligned):
  //   [0, 8 KB)              float2 cw[1024]  = {||c||^2, w}
  //   [8 KB, 136 KB)         bf16 chi[1024*64]
  //   [136 KB, 264 KB)       bf16 clo[1024*64]
  char* ws = (char*)d_ws;
  float2* cw  = (float2*)ws;
  __bf16* chi = (__bf16*)(ws + 8192);
  __bf16* clo = (__bf16*)(ws + 8192 + C_TOTAL * D_DIM * 2);

  const int n = in_sizes[0] / D_DIM;  // rows of x (131072)

  svm_prep_kernel<<<1, 256, 0, stream>>>(centers, coefs, cw, chi, clo);
  svm_main_kernel<<<n / 128, 256, 0, stream>>>(x, cw, chi, clo, mad, out);
}
